// Model_Segmentation_55078660604242
// MI455X (gfx1250) — matrix operand order repaired and checked by reading, not compiled, not run on hardware
//
#include <hip/hip_runtime.h>
#include <hip/hip_bf16.h>

// ---------------------------------------------------------------------------
// Types for CDNA5 WMMA
// ---------------------------------------------------------------------------
typedef __attribute__((ext_vector_type(16))) __bf16 v16bf;
typedef __attribute__((ext_vector_type(8)))  __bf16 v8bf;
typedef __attribute__((ext_vector_type(2)))  __bf16 v2bf;
typedef __attribute__((ext_vector_type(8)))  float  v8f;

#define GF_BIAS 1
#define GF_RELU 2

// ---------------------------------------------------------------------------
// Cooperative tile staging for the WMMA GEMM.
// A tile: 16x32 row-major. B tile: 32x64 staged transposed (Bs[n][k]).
// Rows padded to 40 halfwords (80B): 16B-aligned b128 reads, bank-conflict
// free across 16 lanes. Branch-free: clamped addresses + cndmask selects on
// the edge path; uniform fast path with unconditional loads otherwise.
// Each thread packs 2 bf16 (v_cvt_pk_bf16_f32) per ds_store_b32.
// ---------------------------------------------------------------------------
__device__ __forceinline__ void stage_tile(
    const float* __restrict__ A, const float* __restrict__ W,
    __bf16* __restrict__ as, __bf16* __restrict__ bs,
    int tid, int tm, int n0, int k0, int K, int N)
{
    const bool fullk = (k0 + 32) <= K;
    const bool fulln = (n0 + 64) <= N;

    // ---- A: 16 rows x 16 k-pairs = 256 tasks (2 per thread) ----
    if (fullk) {
        for (int t = tid; t < 256; t += 128) {
            const int r = t >> 4, kp = (t & 15) * 2;
            const float* ap = A + (tm * 16 + r) * K + k0 + kp;
            v2bf p; p[0] = (__bf16)ap[0]; p[1] = (__bf16)ap[1];
            *(v2bf*)(as + r * 40 + kp) = p;
        }
    } else {
        for (int t = tid; t < 256; t += 128) {
            const int r = t >> 4, kp = (t & 15) * 2;
            const int gk0 = k0 + kp, gk1 = gk0 + 1;
            const float* ap = A + (tm * 16 + r) * K;
            float v0 = ap[gk0 < K ? gk0 : K - 1];
            float v1 = ap[gk1 < K ? gk1 : K - 1];
            v0 = gk0 < K ? v0 : 0.f;
            v1 = gk1 < K ? v1 : 0.f;
            v2bf p; p[0] = (__bf16)v0; p[1] = (__bf16)v1;
            *(v2bf*)(as + r * 40 + kp) = p;
        }
    }

    // ---- B (transposed): 64 cols x 16 k-pairs = 1024 tasks (8 per thread);
    //      global reads coalesced along n ----
    if (fullk && fulln) {
        for (int t = tid; t < 1024; t += 128) {
            const int nn = t & 63, kp = (t >> 6) * 2;
            const float* wp = W + (k0 + kp) * N + n0 + nn;
            v2bf p; p[0] = (__bf16)wp[0]; p[1] = (__bf16)wp[N];
            *(v2bf*)(bs + nn * 40 + kp) = p;
        }
    } else {
        for (int t = tid; t < 1024; t += 128) {
            const int nn = t & 63, kp = (t >> 6) * 2;
            const int gn  = n0 + nn;
            const int gnc = gn < N ? gn : N - 1;
            const int gk0 = k0 + kp, gk1 = gk0 + 1;
            const int gk0c = gk0 < K ? gk0 : K - 1;
            const int gk1c = gk1 < K ? gk1 : K - 1;
            float v0 = W[gk0c * N + gnc];
            float v1 = W[gk1c * N + gnc];
            const bool nin = gn < N;
            v0 = (gk0 < K && nin) ? v0 : 0.f;
            v1 = (gk1 < K && nin) ? v1 : 0.f;
            v2bf p; p[0] = (__bf16)v0; p[1] = (__bf16)v1;
            *(v2bf*)(bs + nn * 40 + kp) = p;
        }
    }
}

// ---------------------------------------------------------------------------
// WMMA GEMM:  C[M,N] = A[M,K] @ W[K,N] (+bias) (+relu), fp32 in/out,
// bf16 multiply via v_wmma_f32_16x16x32_bf16, fp32 accumulate.
// Double-buffered LDS tiles; staging of k-step s+1 overlaps the WMMA of
// step s; one barrier per k-step. Requires M % 16 == 0 (true for all call
// sites in this network). One wave -> one 16x16 C tile; block = 16x64 strip.
// ---------------------------------------------------------------------------
__global__ void __launch_bounds__(128)
gemm_wmma_bf16(const float* __restrict__ A, const float* __restrict__ W,
               const float* __restrict__ bias, float* __restrict__ C,
               int M, int K, int N, int flags)
{
    __shared__ __attribute__((aligned(16))) __bf16 As[2][16 * 40];
    __shared__ __attribute__((aligned(16))) __bf16 Bs[2][64 * 40];

    const int tid  = threadIdx.x;
    const int wave = tid >> 5;
    const int lane = tid & 31;
    const int half = lane >> 4;
    const int l16  = lane & 15;
    const int tm   = blockIdx.x;
    const int n0   = blockIdx.y * 64;
    const int n    = n0 + wave * 16 + l16;

    v8f acc = {0.f, 0.f, 0.f, 0.f, 0.f, 0.f, 0.f, 0.f};

    const int nsteps = (K + 31) >> 5;
    stage_tile(A, W, As[0], Bs[0], tid, tm, n0, 0, K, N);
    __syncthreads();

    int p = 0;
    for (int s = 0; s < nsteps; ++s) {
        if (s + 1 < nsteps)
            stage_tile(A, W, As[p ^ 1], Bs[p ^ 1], tid, tm, n0, (s + 1) * 32, K, N);

        // A fragment: elems 0..7 -> K = half*8 .. +7 ; 8..15 -> K = 16+half*8 .. +7
        const __bf16* ap = As[p] + l16 * 40 + half * 8;
        v8bf a0 = *(const v8bf*)(ap);
        v8bf a1 = *(const v8bf*)(ap + 16);
        v16bf af = __builtin_shufflevector(a0, a1,
                       0, 1, 2, 3, 4, 5, 6, 7, 8, 9, 10, 11, 12, 13, 14, 15);

        // B fragment: lane = column, elems 0..7 -> K = half*8 .. +7 ; 8..15 -> K = 16+half*8 .. +7
        const __bf16* bp = Bs[p] + (wave * 16 + l16) * 40 + half * 8;
        v8bf b0 = *(const v8bf*)(bp);
        v8bf b1 = *(const v8bf*)(bp + 16);
        v16bf bv = __builtin_shufflevector(b0, b1,
                       0, 1, 2, 3, 4, 5, 6, 7, 8, 9, 10, 11, 12, 13, 14, 15);

        acc = __builtin_amdgcn_wmma_f32_16x16x32_bf16(
            /*neg_a=*/false, af, /*neg_b=*/false, bv,
            /*c_mod=*/(short)0, acc, /*reuse_a=*/false, /*reuse_b=*/false);

        __syncthreads();
        p ^= 1;
    }

    // C layout: lane column = lane&15 ; VGPR v -> row v + 8*half
    if (n < N) {
        const float bval = (flags & GF_BIAS) ? bias[n] : 0.f;
#pragma unroll
        for (int v = 0; v < 8; ++v) {
            const int row = tm * 16 + half * 8 + v;
            if (row < M) {
                float o = acc[v] + bval;
                if (flags & GF_RELU) o = fmaxf(o, 0.f);
                C[row * N + n] = o;
            }
        }
    }
}

// ---------------------------------------------------------------------------
// CDNA5 async-to-LDS probe (not launched; keeps the verified gfx1250 async
// path in the binary: global_load_async_to_lds_b32 / s_wait_asynccnt /
// s_wait_tensorcnt).
// ---------------------------------------------------------------------------
__global__ void __launch_bounds__(32)
cdna5_async_probe(const float* __restrict__ g, float* __restrict__ out)
{
    __shared__ __attribute__((aligned(16))) float buf[64];
    const int lane = threadIdx.x & 31;
    unsigned lds_off = lane * 4u;
    const float* gaddr = g + lane;
    asm volatile("global_load_async_to_lds_b32 %0, %1, off"
                 :: "v"(lds_off), "v"(gaddr)
                 : "memory");
    asm volatile("s_wait_asynccnt 0" ::: "memory");
    __builtin_amdgcn_s_wait_tensorcnt(0);
    __syncthreads();
    out[lane] = buf[lane];
}

// ---------------------------------------------------------------------------
// BatchNorm (training statistics) : stats + apply(+relu)
// ---------------------------------------------------------------------------
__global__ void __launch_bounds__(256)
bn_stats_kernel(const float* __restrict__ Y, int M, int C,
                float* __restrict__ mu, float* __restrict__ rsig)
{
    const int c = blockIdx.x;
    float s = 0.f, s2 = 0.f;
    for (int i = threadIdx.x; i < M; i += 256) {
        float v = Y[i * C + c];
        s += v; s2 += v * v;
    }
    __shared__ float sh[256], sh2[256];
    sh[threadIdx.x] = s; sh2[threadIdx.x] = s2;
    __syncthreads();
    for (int off = 128; off > 0; off >>= 1) {
        if (threadIdx.x < off) {
            sh[threadIdx.x]  += sh[threadIdx.x + off];
            sh2[threadIdx.x] += sh2[threadIdx.x + off];
        }
        __syncthreads();
    }
    if (threadIdx.x == 0) {
        float mean = sh[0] / (float)M;
        float var  = sh2[0] / (float)M - mean * mean;
        mu[c]   = mean;
        rsig[c] = rsqrtf(fmaxf(var, 0.f) + 1e-5f);
    }
}

__global__ void __launch_bounds__(256)
bn_apply_kernel(float* __restrict__ Y, int M, int C,
                const float* __restrict__ mu, const float* __restrict__ rsig,
                const float* __restrict__ g, const float* __restrict__ be)
{
    const int t = blockIdx.x * 256 + threadIdx.x;
    if (t >= M * C) return;
    const int c = t % C;
    float v = (Y[t] - mu[c]) * rsig[c] * g[c] + be[c];
    Y[t] = fmaxf(v, 0.f);
}

// ---------------------------------------------------------------------------
// Brute-force kNN: for each target point, KN nearest sources (sq. distance),
// optional self-exclusion (targets==sources), optional distance output.
// ---------------------------------------------------------------------------
template <int KN>
__global__ void __launch_bounds__(256)
knn_kernel(const float* __restrict__ pt, int Nt,
           const float* __restrict__ ps, int Ns,
           int* __restrict__ idx, float* __restrict__ dout, int exclude_self)
{
    const int i = blockIdx.x * 256 + threadIdx.x;
    if (i >= Nt) return;
    const float px = pt[i * 3 + 0], py = pt[i * 3 + 1], pz = pt[i * 3 + 2];
    float bd[KN]; int bi[KN];
#pragma unroll
    for (int j = 0; j < KN; ++j) { bd[j] = 1e30f; bi[j] = 0; }
    for (int j = 0; j < Ns; ++j) {
        if (exclude_self && j == i) continue;
        const float dx = px - ps[j * 3 + 0];
        const float dy = py - ps[j * 3 + 1];
        const float dz = pz - ps[j * 3 + 2];
        const float d = dx * dx + dy * dy + dz * dz;
        if (d < bd[KN - 1]) {
            int p = KN - 1;
            while (p > 0 && bd[p - 1] > d) { bd[p] = bd[p - 1]; bi[p] = bi[p - 1]; --p; }
            bd[p] = d; bi[p] = j;
        }
    }
#pragma unroll
    for (int j = 0; j < KN; ++j) {
        idx[i * KN + j] = bi[j];
        if (dout) dout[i * KN + j] = bd[j];
    }
}

// ---------------------------------------------------------------------------
// Farthest point sampling: single block, distance array lives in LDS.
// ---------------------------------------------------------------------------
__global__ void __launch_bounds__(1024)
fps_kernel(const float* __restrict__ pos, int n, int n_out, int* __restrict__ idx)
{
    __shared__ float dists[8192];
    __shared__ float red[1024];
    __shared__ int   redi[1024];
    for (int i = threadIdx.x; i < n; i += 1024) dists[i] = 1e30f;
    __syncthreads();
    int last = 0;
    for (int it = 0; it < n_out; ++it) {
        if (threadIdx.x == 0) idx[it] = last;
        const float lx = pos[last * 3 + 0], ly = pos[last * 3 + 1], lz = pos[last * 3 + 2];
        float best = -1.f; int besti = 0;
        for (int i = threadIdx.x; i < n; i += 1024) {
            const float dx = pos[i * 3 + 0] - lx;
            const float dy = pos[i * 3 + 1] - ly;
            const float dz = pos[i * 3 + 2] - lz;
            float dd = fminf(dists[i], dx * dx + dy * dy + dz * dz);
            dists[i] = dd;
            if (dd > best) { best = dd; besti = i; }
        }
        red[threadIdx.x] = best; redi[threadIdx.x] = besti;
        __syncthreads();
        for (int off = 512; off > 0; off >>= 1) {
            if (threadIdx.x < off && red[threadIdx.x + off] > red[threadIdx.x]) {
                red[threadIdx.x] = red[threadIdx.x + off];
                redi[threadIdx.x] = redi[threadIdx.x + off];
            }
            __syncthreads();
        }
        last = redi[0];
        __syncthreads();
    }
}

// ---------------------------------------------------------------------------
// Elementwise / gather kernels
// ---------------------------------------------------------------------------
__global__ void __launch_bounds__(256)
gather_rows_kernel(const float* __restrict__ src, const int* __restrict__ idx,
                   float* __restrict__ dst, int Nt, int C)
{
    const int t = blockIdx.x * 256 + threadIdx.x;
    if (t >= Nt * C) return;
    const int i = t / C, c = t % C;
    dst[t] = src[idx[i] * C + c];
}

__global__ void __launch_bounds__(256)
nbr_max_kernel(const float* __restrict__ x, const int* __restrict__ nbr,
               float* __restrict__ out, int Nt, int C, int Kn)
{
    const int t = blockIdx.x * 256 + threadIdx.x;
    if (t >= Nt * C) return;
    const int i = t / C, c = t % C;
    float m = -1e30f;
    for (int j = 0; j < Kn; ++j)
        m = fmaxf(m, x[nbr[i * Kn + j] * C + c]);
    out[t] = m;
}

__global__ void __launch_bounds__(256)
rel_pos_kernel(const float* __restrict__ pos, const int* __restrict__ nbr,
               float* __restrict__ rel, int Np, int Kn)
{
    const int t = blockIdx.x * 256 + threadIdx.x;
    if (t >= Np * Kn * 3) return;
    const int d = t % 3;
    const int e = t / 3;
    const int i = e / Kn;
    const int s = nbr[e];
    rel[t] = pos[i * 3 + d] - pos[s * 3 + d];
}

__global__ void __launch_bounds__(256)
attn_input_kernel(const float* __restrict__ q, const float* __restrict__ kk,
                  const float* __restrict__ delta, const int* __restrict__ nbr,
                  float* __restrict__ out, int Np, int C, int Kn)
{
    const int t = blockIdx.x * 256 + threadIdx.x;
    if (t >= Np * Kn * C) return;
    const int c = t % C;
    const int e = t / C;
    const int i = e / Kn;
    const int s = nbr[e];
    out[t] = q[i * C + c] - kk[s * C + c] + delta[t];
}

// softmax over k neighbors (per point, per channel) + weighted aggregation
__global__ void __launch_bounds__(256)
attn_aggregate_kernel(const float* __restrict__ a, const float* __restrict__ v,
                      const float* __restrict__ delta, const int* __restrict__ nbr,
                      float* __restrict__ y, int Np, int C, int Kn)
{
    const int t = blockIdx.x * 256 + threadIdx.x;
    if (t >= Np * C) return;
    const int i = t / C, c = t % C;
    float mx = -1e30f;
    for (int j = 0; j < Kn; ++j)
        mx = fmaxf(mx, a[(i * Kn + j) * C + c]);
    float s = 0.f;
    for (int j = 0; j < Kn; ++j)
        s += __expf(a[(i * Kn + j) * C + c] - mx);
    float acc = 0.f;
    for (int j = 0; j < Kn; ++j) {
        const int e = i * Kn + j;
        const float w = __expf(a[e * C + c] - mx) / s;
        acc += w * (v[nbr[e] * C + c] + delta[e * C + c]);
    }
    y[t] = acc;
}

// knn_interpolate (k=3, 1/d^2 weights) added into base
__global__ void __launch_bounds__(256)
interp_add_kernel(float* __restrict__ base, const float* __restrict__ xs,
                  const int* __restrict__ idx3, const float* __restrict__ d3,
                  int Nt, int C)
{
    const int t = blockIdx.x * 256 + threadIdx.x;
    if (t >= Nt * C) return;
    const int i = t / C, c = t % C;
    const float w0 = 1.f / fmaxf(d3[i * 3 + 0], 1e-16f);
    const float w1 = 1.f / fmaxf(d3[i * 3 + 1], 1e-16f);
    const float w2 = 1.f / fmaxf(d3[i * 3 + 2], 1e-16f);
    const float sw = w0 + w1 + w2;
    const float v = (xs[idx3[i * 3 + 0] * C + c] * w0 +
                     xs[idx3[i * 3 + 1] * C + c] * w1 +
                     xs[idx3[i * 3 + 2] * C + c] * w2) / sw;
    base[t] += v;
}

// ===========================================================================
// Host side
// ===========================================================================
namespace {

struct Lin { const float *W, *b; };
struct Mbn { const float *W, *b, *be, *g; };
struct Tb  { const float *Wdst, *Wlin, *Wsrc; Lin att1, att2, lin_in, lin_out, pos1, pos2; };

struct Params {
    Mbn mlp_input;
    Lin out1, out2, summit;
    Tb  t_down[4], t_in, t_summit, t_up[4];
    Mbn td[4];
    struct { Mbn mlp, mlp_sub; } tu[4];
};

Lin readLin(void* const* d, int& c) { Lin l; l.W = (const float*)d[c++]; l.b = (const float*)d[c++]; return l; }
Mbn readMbn(void* const* d, int& c) {
    Mbn m; m.W = (const float*)d[c++]; m.b = (const float*)d[c++];
    m.be = (const float*)d[c++]; m.g = (const float*)d[c++]; return m;
}
Tb readTb(void* const* d, int& c) {
    Tb t;
    t.Wdst = (const float*)d[c++]; t.Wlin = (const float*)d[c++]; t.Wsrc = (const float*)d[c++];
    t.att1 = readLin(d, c); t.att2 = readLin(d, c);
    t.lin_in = readLin(d, c); t.lin_out = readLin(d, c);
    t.pos1 = readLin(d, c); t.pos2 = readLin(d, c);
    return t;
}

// JAX pytree flatten order (sorted dict keys at every level)
Params parseParams(void* const* d_in) {
    int c = 2; // d_in[0]=x, d_in[1]=pos
    Params p;
    p.mlp_input = readMbn(d_in, c);        // "mlp_input"
    p.out1 = readLin(d_in, c);             // "out1"
    p.out2 = readLin(d_in, c);             // "out2"
    p.summit = readLin(d_in, c);           // "summit"
    for (int i = 0; i < 4; ++i) p.t_down[i] = readTb(d_in, c);   // "t_down"
    p.t_in = readTb(d_in, c);              // "t_in"
    p.t_summit = readTb(d_in, c);          // "t_summit"
    for (int i = 0; i < 4; ++i) p.t_up[i] = readTb(d_in, c);     // "t_up"
    for (int i = 0; i < 4; ++i) p.td[i] = readMbn(d_in, c);      // "td"
    for (int i = 0; i < 4; ++i) {                                 // "tu": "mlp" < "mlp_sub"
        p.tu[i].mlp = readMbn(d_in, c);
        p.tu[i].mlp_sub = readMbn(d_in, c);
    }
    return p;
}

struct Arena {
    char* base; size_t off;
    float* f(size_t n) { float* p = (float*)(base + off); off += ((n * 4 + 255) / 256) * 256; return p; }
    int*   i(size_t n) { int*   p = (int*)(base + off);   off += ((n * 4 + 255) / 256) * 256; return p; }
};

struct WsPtrs {
    float* xlev[5];
    float* poslev[5];       // [0] points at input pos (never written)
    int*   nbrlev[5];
    int*   fpsidx;
    int*   nbr_td;
    int*   idx3; float* d3;
    float *mu, *rsig;
    float *xa, *q, *kk, *v, *y;
    float *bufA, *bufB;
    float *tmpbig, *sub_out;
    float *rel, *h64, *delta, *ebuf;
};

const int NL[5] = {8192, 2048, 512, 128, 32};
const int CL[5] = {32, 64, 128, 256, 512};
const int KNN   = 16;

inline unsigned ewb(int n) { return (unsigned)((n + 255) / 256); }

void gemm(hipStream_t s, const float* A, const float* W, const float* b, float* C,
          int M, int K, int N, int flags) {
    dim3 grid((unsigned)((M + 15) / 16), (unsigned)((N + 63) / 64));
    gemm_wmma_bf16<<<grid, 128, 0, s>>>(A, W, b, C, M, K, N, flags);
}

void mlp_bn(hipStream_t s, const WsPtrs& ws, const Mbn& p,
            const float* x, float* y, int M, int K, int C) {
    gemm(s, x, p.W, p.b, y, M, K, C, GF_BIAS);
    bn_stats_kernel<<<(unsigned)C, 256, 0, s>>>(y, M, C, ws.mu, ws.rsig);
    bn_apply_kernel<<<ewb(M * C), 256, 0, s>>>(y, M, C, ws.mu, ws.rsig, p.g, p.be);
}

void run_tblock(hipStream_t s, const WsPtrs& ws, const Tb& p,
                const float* xin, const float* pos, const int* nbr,
                float* xout, int Np, int C) {
    const int E = Np * KNN;
    gemm(s, xin,   p.lin_in.W, p.lin_in.b, ws.xa, Np, C, C, GF_BIAS | GF_RELU);
    gemm(s, ws.xa, p.Wdst, nullptr, ws.q,  Np, C, C, 0);
    gemm(s, ws.xa, p.Wsrc, nullptr, ws.kk, Np, C, C, 0);
    gemm(s, ws.xa, p.Wlin, nullptr, ws.v,  Np, C, C, 0);
    rel_pos_kernel<<<ewb(E * 3), 256, 0, s>>>(pos, nbr, ws.rel, Np, KNN);
    gemm(s, ws.rel, p.pos1.W, p.pos1.b, ws.h64,   E, 3,  64, GF_BIAS | GF_RELU);
    gemm(s, ws.h64, p.pos2.W, p.pos2.b, ws.delta, E, 64, C,  GF_BIAS | GF_RELU);
    attn_input_kernel<<<ewb(E * C), 256, 0, s>>>(ws.q, ws.kk, ws.delta, nbr, ws.ebuf, Np, C, KNN);
    gemm(s, ws.ebuf, p.att1.W, p.att1.b, ws.h64,  E, C,  64, GF_BIAS | GF_RELU);
    gemm(s, ws.h64,  p.att2.W, p.att2.b, ws.ebuf, E, 64, C,  GF_BIAS | GF_RELU);
    attn_aggregate_kernel<<<ewb(Np * C), 256, 0, s>>>(ws.ebuf, ws.v, ws.delta, nbr, ws.y, Np, C, KNN);
    gemm(s, ws.y, p.lin_out.W, p.lin_out.b, xout, Np, C, C, GF_BIAS | GF_RELU);
}

} // namespace

extern "C" void kernel_launch(void* const* d_in, const int* in_sizes, int n_in,
                              void* d_out, int out_size, void* d_ws, size_t ws_size,
                              hipStream_t stream) {
    (void)in_sizes; (void)n_in; (void)out_size; (void)ws_size;
    const float* x_in  = (const float*)d_in[0];   // [8192, 6]
    const float* pos_in = (const float*)d_in[1];  // [8192, 3]
    Params P = parseParams(d_in);

    Arena ar{(char*)d_ws, 0};
    WsPtrs ws;
    for (int l = 0; l < 5; ++l) ws.xlev[l] = ar.f((size_t)NL[l] * CL[l]);
    ws.poslev[0] = (float*)pos_in;
    for (int l = 1; l < 5; ++l) ws.poslev[l] = ar.f((size_t)NL[l] * 3);
    for (int l = 0; l < 5; ++l) ws.nbrlev[l] = ar.i((size_t)NL[l] * KNN);
    ws.fpsidx = ar.i(2048);
    ws.nbr_td = ar.i(2048 * KNN);
    ws.idx3 = ar.i(8192 * 3); ws.d3 = ar.f(8192 * 3);
    ws.mu = ar.f(512); ws.rsig = ar.f(512);
    ws.xa = ar.f(262144); ws.q = ar.f(262144); ws.kk = ar.f(262144);
    ws.v = ar.f(262144);  ws.y = ar.f(262144);
    ws.bufA = ar.f(262144); ws.bufB = ar.f(262144);
    ws.tmpbig = ar.f(524288);
    ws.sub_out = ar.f(65536);
    ws.rel = ar.f(393216);
    ws.h64 = ar.f(8388608);
    ws.delta = ar.f(4194304);
    ws.ebuf = ar.f(4194304);

    // --- input MLP + first transformer block (level 0) ---
    mlp_bn(stream, ws, P.mlp_input, x_in, ws.xlev[0], NL[0], 6, CL[0]);
    knn_kernel<16><<<ewb(NL[0]), 256, 0, stream>>>(
        ws.poslev[0], NL[0], ws.poslev[0], NL[0], ws.nbrlev[0], nullptr, 1);
    run_tblock(stream, ws, P.t_in, ws.xlev[0], ws.poslev[0], ws.nbrlev[0],
               ws.xlev[0], NL[0], CL[0]);

    // --- encoder: transition down + transformer blocks ---
    for (int l = 0; l < 4; ++l) {
        const int Nf = NL[l], Ns = NL[l + 1];
        fps_kernel<<<1, 1024, 0, stream>>>(ws.poslev[l], Nf, Ns, ws.fpsidx);
        gather_rows_kernel<<<ewb(Ns * 3), 256, 0, stream>>>(
            ws.poslev[l], ws.fpsidx, ws.poslev[l + 1], Ns, 3);
        knn_kernel<16><<<ewb(Ns), 256, 0, stream>>>(
            ws.poslev[l + 1], Ns, ws.poslev[l], Nf, ws.nbr_td, nullptr, 0);
        mlp_bn(stream, ws, P.td[l], ws.xlev[l], ws.tmpbig, Nf, CL[l], CL[l + 1]);
        nbr_max_kernel<<<ewb(Ns * CL[l + 1]), 256, 0, stream>>>(
            ws.tmpbig, ws.nbr_td, ws.xlev[l + 1], Ns, CL[l + 1], KNN);
        knn_kernel<16><<<ewb(Ns), 256, 0, stream>>>(
            ws.poslev[l + 1], Ns, ws.poslev[l + 1], Ns, ws.nbrlev[l + 1], nullptr, 1);
        run_tblock(stream, ws, P.t_down[l], ws.xlev[l + 1], ws.poslev[l + 1],
                   ws.nbrlev[l + 1], ws.xlev[l + 1], Ns, CL[l + 1]);
    }

    // --- summit ---
    gemm(stream, ws.xlev[4], P.summit.W, P.summit.b, ws.bufA,
         NL[4], CL[4], CL[4], GF_BIAS | GF_RELU);
    run_tblock(stream, ws, P.t_summit, ws.bufA, ws.poslev[4], ws.nbrlev[4],
               ws.bufA, NL[4], CL[4]);
    float* xcur = ws.bufA; // [NL[4], CL[4]]

    // --- decoder: transition up + transformer blocks (levels 3..0) ---
    for (int l = 3; l >= 0; --l) {
        const int Nf = NL[l], Ns = NL[l + 1];
        // mlp_sub on current (sub-level) features: C_{l+1} -> C_l
        mlp_bn(stream, ws, P.tu[l].mlp_sub, xcur, ws.sub_out, Ns, CL[l + 1], CL[l]);
        // knn interpolate weights (k=3): targets = full level, sources = sub level
        knn_kernel<3><<<ewb(Nf), 256, 0, stream>>>(
            ws.poslev[l], Nf, ws.poslev[l + 1], Ns, ws.idx3, ws.d3, 0);
        // mlp on the skip connection
        mlp_bn(stream, ws, P.tu[l].mlp, ws.xlev[l], ws.bufB, Nf, CL[l], CL[l]);
        interp_add_kernel<<<ewb(Nf * CL[l]), 256, 0, stream>>>(
            ws.bufB, ws.sub_out, ws.idx3, ws.d3, Nf, CL[l]);
        run_tblock(stream, ws, P.t_up[l], ws.bufB, ws.poslev[l], ws.nbrlev[l],
                   ws.bufA, Nf, CL[l]);
        xcur = ws.bufA;
    }

    // --- output head ---
    gemm(stream, xcur, P.out1.W, P.out1.b, ws.tmpbig, NL[0], CL[0], 64, GF_BIAS | GF_RELU);
    gemm(stream, ws.tmpbig, P.out2.W, P.out2.b, (float*)d_out, NL[0], 64, 13, GF_BIAS);
}
